// TemporalAggregator_29867202576397
// MI455X (gfx1250) — compile-verified, matrix operand order checked by reading, not run on hardware
//
#include <hip/hip_runtime.h>

// ---- problem constants (from reference) ----
#define Bk  8
#define Tk  32
#define Gk  32
#define Ek  768
#define Hk  12
#define Lk  4
#define Sk  (Tk*Gk)        // 1024
#define HDk (Ek/Hk)        // 64
#define Mk  (Bk*Sk)        // 8192
#define LD3E (3*Ek)        // 2304

typedef __bf16 bf16;
typedef __attribute__((ext_vector_type(16))) __bf16 bf16x16;
typedef __attribute__((ext_vector_type(8)))  __bf16 bf16x8;
typedef __attribute__((ext_vector_type(8)))  float  floatx8;

// Load one 16-lane-row fragment (16x32 A-matrix / 32x16 B-matrix, bf16).
// Lane row = lane&15, hv = lane>>4. K split per ISA: half 0 -> K{0..7,16..23},
// half 1 -> K{8..15,24..31}; two contiguous 16B loads per lane.
static __device__ inline bf16x16 frag_load(const bf16* __restrict__ p, int hv) {
    bf16x8 lo = *(const bf16x8*)(p + hv * 8);
    bf16x8 hi = *(const bf16x8*)(p + 16 + hv * 8);
    bf16x16 r;
#pragma unroll
    for (int i = 0; i < 8; ++i) { r[i] = lo[i]; r[i + 8] = hi[i]; }
    return r;
}

static __device__ inline floatx8 wmma_bf16(bf16x16 a, bf16x16 b, floatx8 c) {
    return __builtin_amdgcn_wmma_f32_16x16x32_bf16(false, a, false, b, (short)0, c,
                                                   false, false);
}

static __device__ inline float gelu_new(float x) {
    const float c = 0.7978845608028654f;
    return 0.5f * x * (1.0f + tanhf(c * (x + 0.044715f * x * x * x)));
}

// ---------------- fp32 -> bf16 weight conversion ----------------
__global__ void cvt_bf16_kernel(const float* __restrict__ in, bf16* __restrict__ out,
                                int n) {
    int i = blockIdx.x * 256 + threadIdx.x;
    if (i < n) out[i] = (bf16)in[i];
}

// ---------------- embedding: h = x + wpe[clip(date)] ----------------
__global__ __launch_bounds__(256) void embed_kernel(const float* __restrict__ x,
                                                    const int* __restrict__ dates,
                                                    const float* __restrict__ wpe,
                                                    float* __restrict__ h) {
    int row = blockIdx.x;               // b*S + s
    int b = row / Sk, s = row - b * Sk, t = s / Gk;
    int d = dates[b * Tk + t];
    d = d < 0 ? 0 : (d > 49 ? 49 : d);
    const float* xr = x + (size_t)row * Ek;
    const float* pr = wpe + (size_t)d * Ek;
    float* hr = h + (size_t)row * Ek;
    for (int e = threadIdx.x; e < Ek; e += 256) hr[e] = xr[e] + pr[e];
}

// ---------------- LayerNorm (row per block); optional bf16/f32 outputs -------
__global__ __launch_bounds__(256) void ln_kernel(const float* __restrict__ in,
                                                 const float* __restrict__ w,
                                                 const float* __restrict__ bv,
                                                 bf16* __restrict__ outbf,
                                                 float* __restrict__ outf) {
    int row = blockIdx.x;
    int tid = threadIdx.x, lane = tid & 31, wid = tid >> 5;
    const float* p = in + (size_t)row * Ek;
    float v[3];
    float s = 0.f, s2 = 0.f;
    int i = 0;
    for (int e = tid; e < Ek; e += 256, ++i) {
        float xv = p[e];
        v[i] = xv; s += xv; s2 += xv * xv;
    }
#pragma unroll
    for (int off = 16; off > 0; off >>= 1) {
        s  += __shfl_down(s,  off, 32);
        s2 += __shfl_down(s2, off, 32);
    }
    __shared__ float rs[8], rs2[8];
    if (lane == 0) { rs[wid] = s; rs2[wid] = s2; }
    __syncthreads();
    if (tid == 0) {
        float a = 0.f, b = 0.f;
        for (int k = 0; k < 8; ++k) { a += rs[k]; b += rs2[k]; }
        rs[0] = a; rs2[0] = b;
    }
    __syncthreads();
    float mean = rs[0] * (1.0f / Ek);
    float var  = rs2[0] * (1.0f / Ek) - mean * mean;
    float rstd = rsqrtf(var + 1e-5f);
    i = 0;
    for (int e = tid; e < Ek; e += 256, ++i) {
        float y = (v[i] - mean) * rstd * w[e] + bv[e];
        if (outbf) outbf[(size_t)row * Ek + e] = (bf16)y;
        if (outf)  outf[(size_t)row * Ek + e]  = y;
    }
}

// ---------------- WMMA GEMM: C[M,N] = A[M,K] * W[N,K]^T + bias (+act)(+res) --
// block = 256 threads = 8 waves as 2(m) x 4(n); wave tile 32 x 64 (8 WMMA per
// k-step on 6 fragment loads); block tile 64 x 256. Unroll-by-2 double
// buffering: two fragment register sets alternate, so loads for one set issue
// while WMMAs consume the other -- no rotation copies, partial loadcnt waits.
// Requires K % 64 == 0 and K >= 64 (true here: 768, 3072).
__global__ __launch_bounds__(256) void gemm_bf16_kernel(
    const bf16* __restrict__ A, const bf16* __restrict__ W,
    const float* __restrict__ bias, const float* __restrict__ resid,
    float* __restrict__ outf, bf16* __restrict__ outbf,
    int N, int K, int act) {
    int tid = threadIdx.x;
    int wid = tid >> 5, lane = tid & 31;
    int hv = lane >> 4, ln16 = lane & 15;
    int wm = wid >> 2, wn = wid & 3;
    int m0 = blockIdx.x * 64 + wm * 32;
    int n0 = blockIdx.y * 256 + wn * 64;

    floatx8 acc[8];
#pragma unroll
    for (int t = 0; t < 8; ++t) acc[t] = (floatx8)(0.f);

    const bf16* arow0 = A + (size_t)(m0 + ln16) * K;
    const bf16* arow1 = A + (size_t)(m0 + 16 + ln16) * K;
    const bf16* wrow[4];
#pragma unroll
    for (int t = 0; t < 4; ++t) wrow[t] = W + (size_t)(n0 + t * 16 + ln16) * K;

    // set A <- k = 0
    bf16x16 a0A = frag_load(arow0, hv);
    bf16x16 a1A = frag_load(arow1, hv);
    bf16x16 wAf[4];
#pragma unroll
    for (int t = 0; t < 4; ++t) wAf[t] = frag_load(wrow[t], hv);

    bf16x16 a0B, a1B, wBf[4];
    int k0 = 0;
    for (; k0 < K - 64; k0 += 64) {
        // load set B (k0+32) while computing on set A (k0)
        a0B = frag_load(arow0 + k0 + 32, hv);
        a1B = frag_load(arow1 + k0 + 32, hv);
#pragma unroll
        for (int t = 0; t < 4; ++t) wBf[t] = frag_load(wrow[t] + k0 + 32, hv);
#pragma unroll
        for (int t = 0; t < 4; ++t) {
            acc[t]     = wmma_bf16(a0A, wAf[t], acc[t]);
            acc[4 + t] = wmma_bf16(a1A, wAf[t], acc[4 + t]);
        }
        // load set A (k0+64) while computing on set B (k0+32)
        a0A = frag_load(arow0 + k0 + 64, hv);
        a1A = frag_load(arow1 + k0 + 64, hv);
#pragma unroll
        for (int t = 0; t < 4; ++t) wAf[t] = frag_load(wrow[t] + k0 + 64, hv);
#pragma unroll
        for (int t = 0; t < 4; ++t) {
            acc[t]     = wmma_bf16(a0B, wBf[t], acc[t]);
            acc[4 + t] = wmma_bf16(a1B, wBf[t], acc[4 + t]);
        }
    }
    // tail: k0 == K-64; set A holds k0, load set B (K-32), compute both
    a0B = frag_load(arow0 + k0 + 32, hv);
    a1B = frag_load(arow1 + k0 + 32, hv);
#pragma unroll
    for (int t = 0; t < 4; ++t) wBf[t] = frag_load(wrow[t] + k0 + 32, hv);
#pragma unroll
    for (int t = 0; t < 4; ++t) {
        acc[t]     = wmma_bf16(a0A, wAf[t], acc[t]);
        acc[4 + t] = wmma_bf16(a1A, wAf[t], acc[4 + t]);
    }
#pragma unroll
    for (int t = 0; t < 4; ++t) {
        acc[t]     = wmma_bf16(a0B, wBf[t], acc[t]);
        acc[4 + t] = wmma_bf16(a1B, wBf[t], acc[4 + t]);
    }

    // epilogue: C layout -> element (m0 + mt*16 + hv*8 + r, n0 + t*16 + ln16)
#pragma unroll
    for (int mt = 0; mt < 2; ++mt) {
#pragma unroll
        for (int t = 0; t < 4; ++t) {
            int col = n0 + t * 16 + ln16;
            float bcol = bias[col];
#pragma unroll
            for (int r = 0; r < 8; ++r) {
                int row = m0 + mt * 16 + hv * 8 + r;
                float val = acc[mt * 4 + t][r] + bcol;
                if (act) val = gelu_new(val);
                size_t idx = (size_t)row * N + col;
                if (resid) val += resid[idx];
                if (outf)  outf[idx]  = val;
                if (outbf) outbf[idx] = (bf16)val;
            }
        }
    }
}

// ---------------- flash attention, block-causal (G=32) ----------------
// 8 waves/block, one 16-row query tile per wave. Tiles are 16-aligned inside
// 32-wide causal blocks => no per-element masking, key loop ends at (g+1)*32.
__global__ __launch_bounds__(256) void attn_kernel(const bf16* __restrict__ qkv,
                                                   bf16* __restrict__ O) {
    __shared__ __align__(16) bf16 smem[8][2560];  // per wave: P 16x32, Vt 64x32
    int tid = threadIdx.x;
    int wid = tid >> 5, lane = tid & 31;
    int hv = lane >> 4, ln16 = lane & 15;

    int blk = blockIdx.x;             // B*H*8
    int qt8 = blk & 7;
    int bh  = blk >> 3;
    int head = bh % Hk, b = bh / Hk;
    int qt = qt8 * 8 + wid;           // 0..63
    int m0 = qt * 16;                 // query row within sequence
    int kmax = (m0 / Gk + 1) * Gk;    // block-causal limit (multiple of 32)

    const bf16* Qb = qkv + (size_t)b * Sk * LD3E + head * HDk;
    const bf16* Kb = Qb + Ek;
    const bf16* Vb = Qb + 2 * Ek;

    bf16* Pl = &smem[wid][0];         // [16][32]
    bf16* Vt = &smem[wid][512];       // [64][32] (d-major, transposed V chunk)

    bf16x16 qf0 = frag_load(Qb + (size_t)(m0 + ln16) * LD3E,      hv);
    bf16x16 qf1 = frag_load(Qb + (size_t)(m0 + ln16) * LD3E + 32, hv);

    floatx8 acc[4];
#pragma unroll
    for (int t = 0; t < 4; ++t) acc[t] = (floatx8)(0.f);
    float mr[8], lr[8];
#pragma unroll
    for (int r = 0; r < 8; ++r) { mr[r] = -1e30f; lr[r] = 0.f; }

    for (int kb = 0; kb < kmax; kb += 32) {
        // stage V chunk transposed: lane owns key row (kb+lane)
        const bf16* vr = Vb + (size_t)(kb + lane) * LD3E;
#pragma unroll
        for (int dv = 0; dv < 8; ++dv) {
            bf16x8 ch = *(const bf16x8*)(vr + dv * 8);
#pragma unroll
            for (int j = 0; j < 8; ++j) Vt[(dv * 8 + j) * 32 + lane] = ch[j];
        }
        // scores: two 16x16 tiles (keys kb..+15, kb+16..+31), accum over d
        floatx8 s0 = (floatx8)(0.f), s1 = (floatx8)(0.f);
        {
            const bf16* kr = Kb + (size_t)(kb + ln16) * LD3E;
            s0 = wmma_bf16(qf0, frag_load(kr,      hv), s0);
            s0 = wmma_bf16(qf1, frag_load(kr + 32, hv), s0);
            const bf16* kr2 = Kb + (size_t)(kb + 16 + ln16) * LD3E;
            s1 = wmma_bf16(qf0, frag_load(kr2,      hv), s1);
            s1 = wmma_bf16(qf1, frag_load(kr2 + 32, hv), s1);
        }
        float alpha[8], p0[8], p1[8];
#pragma unroll
        for (int r = 0; r < 8; ++r) {
            float a0 = s0[r] * 0.125f;   // 1/sqrt(64)
            float a1 = s1[r] * 0.125f;
            float cm = fmaxf(a0, a1);
#pragma unroll
            for (int off = 8; off > 0; off >>= 1)
                cm = fmaxf(cm, __shfl_xor(cm, off, 32));
            float mn = fmaxf(mr[r], cm);
            alpha[r] = __expf(mr[r] - mn);
            mr[r] = mn;
            float e0 = __expf(a0 - mn), e1 = __expf(a1 - mn);
            p0[r] = e0; p1[r] = e1;
            float rsum = e0 + e1;
#pragma unroll
            for (int off = 8; off > 0; off >>= 1)
                rsum += __shfl_xor(rsum, off, 32);
            lr[r] = lr[r] * alpha[r] + rsum;
        }
#pragma unroll
        for (int t = 0; t < 4; ++t)
#pragma unroll
            for (int r = 0; r < 8; ++r) acc[t][r] *= alpha[r];

        // P tile (16 x 32) to LDS in row-major for A-fragment reload
#pragma unroll
        for (int r = 0; r < 8; ++r) {
            Pl[(hv * 8 + r) * 32 + ln16]      = (bf16)p0[r];
            Pl[(hv * 8 + r) * 32 + 16 + ln16] = (bf16)p1[r];
        }
        asm volatile("s_wait_dscnt 0" ::: "memory");  // wave-local LDS ordering

        bf16x16 pf = frag_load(Pl + ln16 * 32, hv);
#pragma unroll
        for (int t = 0; t < 4; ++t) {
            bf16x16 vf = frag_load(Vt + (t * 16 + ln16) * 32, hv);
            acc[t] = wmma_bf16(pf, vf, acc[t]);
        }
    }
    // normalize + store bf16 output [B*S, E]
#pragma unroll
    for (int r = 0; r < 8; ++r) {
        float inv = 1.0f / lr[r];
        size_t row = (size_t)b * Sk + m0 + hv * 8 + r;
#pragma unroll
        for (int t = 0; t < 4; ++t)
            O[row * Ek + head * HDk + t * 16 + ln16] = (bf16)(acc[t][r] * inv);
    }
}

// ---------------- host orchestration ----------------
extern "C" void kernel_launch(void* const* d_in, const int* in_sizes, int n_in,
                              void* d_out, int out_size, void* d_ws, size_t ws_size,
                              hipStream_t stream) {
    (void)in_sizes; (void)n_in; (void)out_size; (void)ws_size;
    const float* x          = (const float*)d_in[0];
    const int*   study_date = (const int*)  d_in[1];
    const float* wpe        = (const float*)d_in[2];
    const float* ln1_w      = (const float*)d_in[3];
    const float* ln1_b      = (const float*)d_in[4];
    const float* in_w       = (const float*)d_in[5];
    const float* in_b       = (const float*)d_in[6];
    const float* out_w      = (const float*)d_in[7];
    const float* out_b      = (const float*)d_in[8];
    const float* ln2_w      = (const float*)d_in[9];
    const float* ln2_b      = (const float*)d_in[10];
    const float* fc_w       = (const float*)d_in[11];
    const float* fc_b       = (const float*)d_in[12];
    const float* proj_w     = (const float*)d_in[13];
    const float* proj_b     = (const float*)d_in[14];
    const float* lnf_w      = (const float*)d_in[15];
    const float* lnf_b      = (const float*)d_in[16];
    float* out = (float*)d_out;

    // workspace layout (all sizes 256B-multiples)
    char* w = (char*)d_ws;
    float* h   = (float*)w;  w += (size_t)Mk * Ek * 4;          // 25.2 MB residual
    bf16* act  = (bf16*)w;   w += (size_t)Mk * Ek * 2;          // 12.6 MB y/o bf16
    bf16* big  = (bf16*)w;   w += (size_t)Mk * 4 * Ek * 2;      // 50.3 MB qkv/fc act
    bf16* wbuf = (bf16*)w;                                      //  4.7 MB weights bf16

    dim3 blk(256);
    embed_kernel<<<Mk, blk, 0, stream>>>(x, study_date, wpe, h);

    for (int l = 0; l < Lk; ++l) {
        // ln1 -> bf16
        ln_kernel<<<Mk, blk, 0, stream>>>(h, ln1_w + l * Ek, ln1_b + l * Ek,
                                          act, nullptr);
        // qkv = y @ in_w^T + in_b  (bf16 out)
        int nIn = 3 * Ek * Ek;
        cvt_bf16_kernel<<<(nIn + 255) / 256, blk, 0, stream>>>(
            in_w + (size_t)l * nIn, wbuf, nIn);
        gemm_bf16_kernel<<<dim3(Mk / 64, (3 * Ek) / 256), blk, 0, stream>>>(
            act, wbuf, in_b + l * 3 * Ek, nullptr, nullptr, big, 3 * Ek, Ek, 0);
        // attention -> act (bf16)
        attn_kernel<<<Bk * Hk * 8, blk, 0, stream>>>(big, act);
        // h += o @ out_w^T + out_b
        int nOut = Ek * Ek;
        cvt_bf16_kernel<<<(nOut + 255) / 256, blk, 0, stream>>>(
            out_w + (size_t)l * nOut, wbuf, nOut);
        gemm_bf16_kernel<<<dim3(Mk / 64, Ek / 256), blk, 0, stream>>>(
            act, wbuf, out_b + l * Ek, h, h, nullptr, Ek, Ek, 0);
        // ln2 -> bf16
        ln_kernel<<<Mk, blk, 0, stream>>>(h, ln2_w + l * Ek, ln2_b + l * Ek,
                                          act, nullptr);
        // g = gelu(y2 @ fc_w^T + fc_b) (bf16 out)
        int nFc = 4 * Ek * Ek;
        cvt_bf16_kernel<<<(nFc + 255) / 256, blk, 0, stream>>>(
            fc_w + (size_t)l * nFc, wbuf, nFc);
        gemm_bf16_kernel<<<dim3(Mk / 64, (4 * Ek) / 256), blk, 0, stream>>>(
            act, wbuf, fc_b + l * 4 * Ek, nullptr, nullptr, big, 4 * Ek, Ek, 1);
        // h += g @ proj_w^T + proj_b
        cvt_bf16_kernel<<<(nFc + 255) / 256, blk, 0, stream>>>(
            proj_w + (size_t)l * nFc, wbuf, nFc);
        gemm_bf16_kernel<<<dim3(Mk / 64, Ek / 256), blk, 0, stream>>>(
            big, wbuf, proj_b + l * Ek, h, h, nullptr, Ek, 4 * Ek, 0);
    }
    // final LN -> fp32 output
    ln_kernel<<<Mk, blk, 0, stream>>>(h, lnf_w, lnf_b, nullptr, out);
}
